// MIC_14413910245450
// MI455X (gfx1250) — matrix operand order repaired and checked by reading, not compiled, not run on hardware
//
#include <hip/hip_runtime.h>

typedef __attribute__((ext_vector_type(16))) _Float16 v16h;
typedef __attribute__((ext_vector_type(8)))  _Float16 v8h;
typedef __attribute__((ext_vector_type(8)))  float    v8f;

#define BDIM 32
#define LDIM 1024
#define CDIM 512

// ---------------------------------------------------------------------------
// Templated LDS-staged WMMA GEMM: D[M,N] = act( gatherA[M,K]*gatherB[K,N]+bias )
// Block = 8 waves; block owns one 16-row M tile and a 512-column N group.
// Per 32-wide K step the block cooperatively stages A (16x32 f16) and B
// (32x512 f16) into LDS with all guards folded into the fill; the math loop is
// then pure ds_load_b128 fragment reads + 4 v_wmma_f32_16x16x32_f16 per wave.
// Requires M % 16 == 0 (all call sites satisfy this).
// ---------------------------------------------------------------------------
struct GP {
  int M, N, K;
  const float* Ab;
  const float* Bb;
  const float* bias;   // per-column bias or nullptr
  float* D;
  int dRow;            // D row stride (col stride == 1)
  // AM==0 (strided): idx = (row/aP)*aS1 + (row%aP)*aS2 + k*aSk
  int aP, aS1, aS2, aSk;
  // AM==1 (conv patch): b=row/an, t=row%an, ci=k/ak, j=k%ak, s=t*astride+j-apad
  //                     idx = b*asb + s*ass + ci*asc  (0 if s outside [0,aslen))
  int an, ak, astride, apad, aslen, asb, ass, asc;
  // AM==2 (branch merge): n=k>>9, c=k&511, idx = n*mgSn + row*512 + c
  int mgSn;
  // BM==0 (strided): idx = k*bSk + col*bSn   BM==2: col*1024 + c*2 + n
  int bSk, bSn;
};

template<int AM, int BM, int ACT>
__global__ __launch_bounds__(256) void wmma_gemm(GP p) {
  __shared__ __align__(16) _Float16 As[16 * 32];     //  1 KB
  __shared__ __align__(16) _Float16 Bs[512 * 32];    // 32 KB
  const int tid  = threadIdx.x;
  const int lane = tid & 31;
  const int wave = tid >> 5;
  const int ngN  = (p.N + 511) >> 9;
  const int bM   = blockIdx.x / ngN;
  const int gN   = blockIdx.x - bM * ngN;
  const int tm   = bM << 4;
  const int ng0  = gN << 9;                 // block's first global column
  const int g  = lane >> 4;
  const int lr = lane & 15;

  // ---- hoisted A-fill decomposition (row fixed per thread) ----------------
  const int fr_l = tid >> 4;                // local A row 0..15
  const int frow = tm + fr_l;               // global A row (< M always)
  const int fkk  = (tid & 15) << 1;         // K offset pair within step
  int aR2 = 0, aBase = 0;
  if constexpr (AM == 0) {
    const int r1 = frow / p.aP; const int r2 = frow - r1 * p.aP;
    aBase = r1 * p.aS1 + r2 * p.aS2;
  } else if constexpr (AM == 1) {
    const int b = frow / p.an; aR2 = frow - b * p.an;
    aBase = b * p.asb;
  } else {
    aBase = frow * CDIM;
  }
  // ---- hoisted B-fill decomposition (2 columns per thread) ----------------
  const int nl0 = tid << 1;                 // local columns nl0, nl0+1
  int cBase[2]; bool cok[2];
#pragma unroll
  for (int r = 0; r < 2; ++r) {
    const int col = ng0 + nl0 + r;
    cok[r] = col < p.N;
    cBase[r] = (BM == 0) ? col * p.bSn : col * (CDIM * 2);
  }

  v8f acc[4] = {{}, {}, {}, {}};
  for (int k0 = 0; k0 < p.K; k0 += 32) {
    __syncthreads();                        // WAR: previous fragment reads done
    // ---- fill A (2 halves -> one packed b32 store) ------------------------
    {
      union { _Float16 h[2]; unsigned u; } pk;
#pragma unroll
      for (int q = 0; q < 2; ++q) {
        const int kk = k0 + fkk + q;
        float f = 0.f;
        if (kk < p.K) {
          if constexpr (AM == 0) {
            f = p.Ab[aBase + kk * p.aSk];
          } else if constexpr (AM == 1) {
            const int ci = kk / p.ak; const int j = kk - ci * p.ak;
            const int s = aR2 * p.astride + j - p.apad;
            if (s >= 0 && s < p.aslen) f = p.Ab[aBase + s * p.ass + ci * p.asc];
          } else {
            const int nn = kk >> 9; const int c = kk & (CDIM - 1);
            f = p.Ab[nn * p.mgSn + aBase + c];
          }
        }
        pk.h[q] = (_Float16)f;
      }
      *(unsigned*)&As[fr_l * 32 + fkk] = pk.u;
    }
    // ---- fill B (2 columns x 32 K, packed pair stores) --------------------
#pragma unroll
    for (int r = 0; r < 2; ++r) {
      const int nl = nl0 + r;
#pragma unroll
      for (int kk2 = 0; kk2 < 32; kk2 += 2) {
        union { _Float16 h[2]; unsigned u; } pk;
#pragma unroll
        for (int q = 0; q < 2; ++q) {
          const int kg = k0 + kk2 + q;
          float f = 0.f;
          if (cok[r] && kg < p.K) {
            if constexpr (BM == 0) f = p.Bb[kg * p.bSk + cBase[r]];
            else { const int nn = kg >> 9; const int c = kg & (CDIM - 1);
                   f = p.Bb[cBase[r] + c * 2 + nn]; }
          }
          pk.h[q] = (_Float16)f;
        }
        *(unsigned*)&Bs[nl * 32 + kk2] = pk.u;
      }
    }
    __syncthreads();
    // ---- A fragment: K = g*16 + i -> two b128 DS reads -----
    union { v16h v; v8h h[2]; } au;
    au.h[0] = *(const v8h*)&As[lr * 32 + (g << 4)];
    au.h[1] = *(const v8h*)&As[lr * 32 + 8 + (g << 4)];
    // ---- 4 B fragments + 4 WMMA ------------------------------------------
#pragma unroll
    for (int j = 0; j < 4; ++j) {
      const int coll = (wave << 6) + (j << 4) + lr;   // local column
      union { v16h v; v8h h[2]; } bu;                 // K = g*16 + i
      const v8h* bp = (const v8h*)&Bs[coll * 32 + (g << 4)];
      bu.h[0] = bp[0];
      bu.h[1] = bp[1];
      acc[j] = __builtin_amdgcn_wmma_f32_16x16x32_f16(false, au.v, false, bu.v,
                                                      (short)0, acc[j],
                                                      false, false);
    }
  }
  // ---- epilogue: D float r -> M = tm + r + 8*g, N = col -------------------
#pragma unroll
  for (int j = 0; j < 4; ++j) {
#pragma unroll
    for (int r = 0; r < 8; ++r) {
      const int mrow = tm + r + (g << 3);
      const int ncol = ng0 + (wave << 6) + (j << 4) + lr;
      if (ncol < p.N) {
        float v = acc[j][r];
        if (p.bias) v += p.bias[ncol];
        if constexpr (ACT == 1) v = tanhf(v);
        else if constexpr (ACT == 2) v = fmaxf(v, 0.f);
        p.D[mrow * p.dRow + ncol] = v;
      }
    }
  }
}

template<int AM, int BM, int ACT>
static inline void launch_gemm(const GP& p, hipStream_t stream) {
  const int tilesM = (p.M + 15) / 16;
  const int ngN = (p.N + 511) / 512;
  wmma_gemm<AM, BM, ACT><<<tilesM * ngN, 256, 0, stream>>>(p);
}

// ---------------------------------------------------------------------------
// Elementwise / reduction kernels
// ---------------------------------------------------------------------------
__global__ void residual_kernel(const float* __restrict__ src,
                                float* __restrict__ res, int dk, int pd) {
  const long long idx = (long long)blockIdx.x * 256 + threadIdx.x;
  if (idx >= (long long)BDIM * LDIM * CDIM) return;
  const int c = (int)(idx & (CDIM - 1));
  const int bl = (int)(idx >> 9);
  const int l = bl & (LDIM - 1);
  const int b = bl >> 10;
  float s = 0.f;
  for (int w = 0; w < dk; ++w) {
    int li = l - pd + w;
    li = li < 0 ? 0 : (li > LDIM - 1 ? LDIM - 1 : li);
    s += src[((long long)(b * LDIM + li)) * CDIM + c];
  }
  res[idx] = src[idx] - s / (float)dk;
}

// x1[(b*C+c)*m + s] : exp-gated even/odd interleave of xe/xo [B*n, C]
__global__ void gate_kernel(const float* __restrict__ xe,
                            const float* __restrict__ xo,
                            float* __restrict__ x1, int n, int m) {
  const int idx = blockIdx.x * 256 + threadIdx.x;
  if (idx >= BDIM * CDIM * m) return;
  const int s = idx % m;
  const int bc = idx / m;
  const int c = bc & (CDIM - 1);
  const int b = bc >> 9;
  const int t = s >> 1;
  const int row = (b * n + t) * CDIM + c;
  const float e = xe[row];
  const float o = xo[row];
  x1[idx] = (s & 1) ? e * expf(o) : o * expf(e);
}

// Cd[s*N2 + u] = cos(2*pi*((s+m-1)*u mod N2)/N2)   (causal-pad DFT, real part)
__global__ void fill_dft(float* Cd, int m, int N2) {
  const int idx = blockIdx.x * 256 + threadIdx.x;
  if (idx >= m * N2) return;
  const int s = idx / N2;
  const int u = idx - s * N2;
  const int ph = ((s + m - 1) * u) % N2;
  Cd[idx] = cosf(6.28318530717958647692f * (float)ph / (float)N2);
}

// Ci[n*m + t] = cos(2*pi*(n*t mod m)/m)/m   (real part of IFFT of real input)
__global__ void fill_idft(float* Ci, int m) {
  const int idx = blockIdx.x * 256 + threadIdx.x;
  if (idx >= m * m) return;
  const int n = idx / m;
  const int t = idx - n * m;
  const int ph = (n * t) % m;
  Ci[idx] = cosf(6.28318530717958647692f * (float)ph / (float)m) / (float)m;
}

__device__ __forceinline__ float block_sum_256(float v) {
  __shared__ float red[256];
  const int tid = threadIdx.x;
  red[tid] = v; __syncthreads();
  for (int s = 128; s > 0; s >>= 1) {
    if (tid < s) red[tid] += red[tid + s];
    __syncthreads();
  }
  const float r = red[0];
  __syncthreads();
  return r;
}

// LN over C of (xr + x1), both stored [(b*C+c)*m + t]; out ypre[(b*m+t)*C + c]
__global__ void ln_mid(const float* __restrict__ xr, const float* __restrict__ x1,
                       const float* __restrict__ gg, const float* __restrict__ bb,
                       float* __restrict__ ypre, int m) {
  const int row = blockIdx.x;           // b*m + t
  const int b = row / m;
  const int t = row - b * m;
  const int c1 = threadIdx.x, c2 = threadIdx.x + 256;
  const int i1 = (b * CDIM + c1) * m + t;
  const int i2 = (b * CDIM + c2) * m + t;
  const float v1 = xr[i1] + x1[i1];
  const float v2 = xr[i2] + x1[i2];
  const float mean = block_sum_256(v1 + v2) * (1.f / CDIM);
  const float d1 = v1 - mean, d2 = v2 - mean;
  const float var = block_sum_256(d1 * d1 + d2 * d2) * (1.f / CDIM);
  const float inv = rsqrtf(var + 1e-5f);
  const int o = row * CDIM;
  ypre[o + c1] = d1 * inv * gg[c1] + bb[c1];
  ypre[o + c2] = d2 * inv * gg[c2] + bb[c2];
}

// tanh(convT)+interp+residual+LN: yt[(b*m+i)*(C*k) + c*k + j], out [B*L, C]
__global__ void ln_out(const float* __restrict__ yt, const float* __restrict__ tb,
                       const float* __restrict__ res,
                       const float* __restrict__ gg, const float* __restrict__ bb,
                       float* __restrict__ outp, int m, int k, int mk) {
  const int row = blockIdx.x;           // b*L + l
  const int b = row >> 10;
  const int l = row & (LDIM - 1);
  const int t = (l * mk) >> 10;         // nearest interp: floor(l*mk/1024)
  const int i = t / k;
  const int j = t - i * k;
  const int base = (b * m + i) * (CDIM * k) + j;
  const int rb = row * CDIM;
  const int c1 = threadIdx.x, c2 = threadIdx.x + 256;
  const float v1 = tanhf(yt[base + c1 * k] + tb[c1]) + res[rb + c1];
  const float v2 = tanhf(yt[base + c2 * k] + tb[c2]) + res[rb + c2];
  const float mean = block_sum_256(v1 + v2) * (1.f / CDIM);
  const float d1 = v1 - mean, d2 = v2 - mean;
  const float var = block_sum_256(d1 * d1 + d2 * d2) * (1.f / CDIM);
  const float inv = rsqrtf(var + 1e-5f);
  outp[rb + c1] = d1 * inv * gg[c1] + bb[c1];
  outp[rb + c2] = d2 * inv * gg[c2] + bb[c2];
}

// final LN of (mg + h2)
__global__ void ln_final(const float* __restrict__ mg, const float* __restrict__ h2,
                         const float* __restrict__ gg, const float* __restrict__ bb,
                         float* __restrict__ outp) {
  const int rb = blockIdx.x * CDIM;
  const int c1 = threadIdx.x, c2 = threadIdx.x + 256;
  const float v1 = mg[rb + c1] + h2[rb + c1];
  const float v2 = mg[rb + c2] + h2[rb + c2];
  const float mean = block_sum_256(v1 + v2) * (1.f / CDIM);
  const float d1 = v1 - mean, d2 = v2 - mean;
  const float var = block_sum_256(d1 * d1 + d2 * d2) * (1.f / CDIM);
  const float inv = rsqrtf(var + 1e-5f);
  outp[rb + c1] = d1 * inv * gg[c1] + bb[c1];
  outp[rb + c2] = d2 * inv * gg[c2] + bb[c2];
}

// ---------------------------------------------------------------------------
extern "C" void kernel_launch(void* const* d_in, const int* in_sizes, int n_in,
                              void* d_out, int out_size, void* d_ws, size_t ws_size,
                              hipStream_t stream) {
  (void)in_sizes; (void)n_in; (void)out_size; (void)ws_size;
  const float* src     = (const float*)d_in[0];
  const float* merge_w = (const float*)d_in[13];
  const float* merge_b = (const float*)d_in[14];
  const float* norm_g  = (const float*)d_in[15];
  const float* norm_b  = (const float*)d_in[16];
  const float* fnn_w1  = (const float*)d_in[17];
  const float* fnn_b1  = (const float*)d_in[18];
  const float* fnn_w2  = (const float*)d_in[19];
  const float* fnn_b2  = (const float*)d_in[20];
  const float* fnn_g   = (const float*)d_in[21];
  const float* fnn_bb  = (const float*)d_in[22];
  float* outp = (float*)d_out;
  float* W = (float*)d_ws;

  const long long BLC = (long long)BDIM * LDIM * CDIM;   // 16,777,216
  float* p_out0 = W;
  float* p_out1 = W + BLC;                 // must be contiguous with out0 (merge)
  float* p_res  = W + 2 * BLC;             // residual; later reused as mg
  float* p_yt   = W + 3 * BLC;             // convT out (max 16,910,336); later h1
  float* p_h2   = p_yt + 16910336;
  float* p_xe   = p_h2 + BLC;
  float* p_xo   = p_xe + 704512;           // B*43*C
  float* p_x1   = p_xo + 704512;
  float* p_xf   = p_x1 + 1392640;          // B*C*85
  float* p_xi   = p_xf + 2768896;          // B*C*169
  float* p_xr   = p_xi + 1392640;
  float* p_yp   = p_xr + 1392640;
  float* p_cd   = p_yp + 1392640;
  float* p_ci   = p_cd + 16384;

  for (int br = 0; br < 2; ++br) {
    const int k  = br ? 24 : 12;
    const int dk = br ? 33 : 17;
    const int n  = br ? 22 : 43;     // strided-conv output length
    const int m  = br ? 43 : 85;     // = 2n-1 (== ISO_K)
    const int N2 = 2 * m - 1;        // causal-padded FFT length
    const int pc = k / 2;
    const int pd = (dk - 1) / 2;
    const int mk = m * k;
    const float* cw = (const float*)d_in[1 + 6 * br];
    const float* cb = (const float*)d_in[2 + 6 * br];
    const float* iw = (const float*)d_in[3 + 6 * br];
    const float* ib = (const float*)d_in[4 + 6 * br];
    const float* tw = (const float*)d_in[5 + 6 * br];
    const float* tb = (const float*)d_in[6 + 6 * br];
    float* p_o = br ? p_out1 : p_out0;

    // series-decomp residual: res = src - moving_avg(src, dk)  [B,L,C]
    residual_kernel<<<(int)((BLC + 255) / 256), 256, 0, stream>>>(src, p_res, dk, pd);

    // even/odd strided conv + tanh as implicit-patch GEMM: [B*n, C] = patch @ cw^T
    for (int par = 0; par < 2; ++par) {
      GP p = {};
      p.M = BDIM * n; p.N = CDIM; p.K = CDIM * k;
      p.Ab = p_res + par * CDIM;             // parity offset folded into base
      p.an = n; p.ak = k; p.astride = k; p.apad = pc; p.aslen = 512;
      p.asb = LDIM * CDIM; p.ass = 2 * CDIM; p.asc = 1;
      p.Bb = cw; p.bSk = 1; p.bSn = CDIM * k;   // w[o][ci][j] transposed
      p.bias = cb; p.D = par ? p_xo : p_xe; p.dRow = CDIM;
      launch_gemm<1, 0, 1>(p, stream);
    }

    gate_kernel<<<(BDIM * CDIM * m + 255) / 256, 256, 0, stream>>>(p_xe, p_xo, p_x1, n, m);

    // real-DFT of causal-padded x1 as GEMM with cosine matrix: xf[B*C, N2]
    fill_dft<<<(m * N2 + 255) / 256, 256, 0, stream>>>(p_cd, m, N2);
    { GP p = {};
      p.M = BDIM * CDIM; p.N = N2; p.K = m;
      p.Ab = p_x1; p.aP = 1 << 30; p.aS1 = 0; p.aS2 = m; p.aSk = 1;
      p.Bb = p_cd; p.bSk = N2; p.bSn = 1;
      p.D = p_xf; p.dRow = N2;
      launch_gemm<0, 0, 0>(p, stream); }

    // isometric conv (kernel length m, valid) + tanh: the big GEMM, K = C*m
    { GP p = {};
      p.M = BDIM * m; p.N = CDIM; p.K = CDIM * m;
      p.Ab = p_xf; p.an = m; p.ak = m; p.astride = 1; p.apad = 0; p.aslen = N2;
      p.asb = CDIM * N2; p.ass = 1; p.asc = N2;
      p.Bb = iw; p.bSk = 1; p.bSn = CDIM * m;
      p.bias = ib; p.D = p_xi; p.dRow = CDIM;   // xi stored [B, m, C]
      launch_gemm<1, 0, 1>(p, stream); }

    // real-IFFT as GEMM: xr[(b,c), t], 1/m folded into matrix
    fill_idft<<<(m * m + 255) / 256, 256, 0, stream>>>(p_ci, m);
    { GP p = {};
      p.M = BDIM * CDIM; p.N = m; p.K = m;
      p.Ab = p_xi; p.aP = CDIM; p.aS1 = m * CDIM; p.aS2 = 1; p.aSk = CDIM;
      p.Bb = p_ci; p.bSk = m; p.bSn = 1;
      p.D = p_xr; p.dRow = m;
      launch_gemm<0, 0, 0>(p, stream); }

    // LN over C of (xr + x1) -> ypre [B, m, C]
    ln_mid<<<BDIM * m, 256, 0, stream>>>(p_xr, p_x1, norm_g, norm_b, p_yp, m);

    // ConvTranspose (stride==k) as GEMM: yt[(b,i), o*k+j], K = C
    { GP p = {};
      p.M = BDIM * m; p.N = CDIM * k; p.K = CDIM;
      p.Ab = p_yp; p.aP = m; p.aS1 = m * CDIM; p.aS2 = CDIM; p.aSk = 1;
      p.Bb = tw; p.bSk = CDIM * k; p.bSn = 1;   // tw[ci][o][j] row-major
      p.D = p_yt; p.dRow = CDIM * k;
      launch_gemm<0, 0, 0>(p, stream); }

    // tanh + bias + nearest interp + residual + LN -> out_br [B*L, C]
    ln_out<<<BDIM * LDIM, 256, 0, stream>>>(p_yt, tb, p_res, norm_g, norm_b,
                                            p_o, m, k, mk);
  }

  // merge: mg[b,l,o] = sum_{n,c} out_n[b,l,c] * merge_w[o,c,n] + merge_b
  float* p_mg = p_res;   // residual buffer is dead now
  { GP p = {};
    p.M = BDIM * LDIM; p.N = CDIM; p.K = 2 * CDIM;
    p.Ab = p_out0; p.mgSn = (int)BLC;
    p.Bb = merge_w; p.bias = merge_b;
    p.D = p_mg; p.dRow = CDIM;
    launch_gemm<2, 2, 0>(p, stream); }

  // FFN (chunked to bound scratch): h = relu(mg@W1+b1); h2 = h@W2+b2
  float* p_h1 = p_yt;
  const int CH = 8192;
  for (int ch = 0; ch < 4; ++ch) {
    { GP p = {};
      p.M = CH; p.N = 4 * CDIM; p.K = CDIM;
      p.Ab = p_mg + (long long)ch * CH * CDIM;
      p.aP = 1 << 30; p.aS1 = 0; p.aS2 = CDIM; p.aSk = 1;
      p.Bb = fnn_w1; p.bSk = 4 * CDIM; p.bSn = 1;
      p.bias = fnn_b1; p.D = p_h1; p.dRow = 4 * CDIM;
      launch_gemm<0, 0, 2>(p, stream); }
    { GP p = {};
      p.M = CH; p.N = CDIM; p.K = 4 * CDIM;
      p.Ab = p_h1; p.aP = 1 << 30; p.aS1 = 0; p.aS2 = 4 * CDIM; p.aSk = 1;
      p.Bb = fnn_w2; p.bSk = CDIM; p.bSn = 1;
      p.bias = fnn_b2; p.D = p_h2 + (long long)ch * CH * CDIM; p.dRow = CDIM;
      launch_gemm<0, 0, 0>(p, stream); }
  }

  ln_final<<<BDIM * LDIM, 256, 0, stream>>>(p_mg, p_h2, fnn_g, fnn_bb, outp);
}
